// LovaszSoftmaxV3_86517821214363
// MI455X (gfx1250) — compile-verified, matrix operand order checked by reading, not run on hardware
//
#include <hip/hip_runtime.h>
#include <math.h>

typedef unsigned int u32;
typedef unsigned long long u64;
typedef __attribute__((ext_vector_type(16))) _Float16 v16h;
typedef __attribute__((ext_vector_type(8)))  float    v8f;

#define NPIX   1179648      // 8*384*384
#define NCH    19
#define HW     147456       // 384*384
#define TILE   2048
#define NBLK   576          // NPIX / TILE
#define IGNORE_IDX (-100)

// ---------- per-pixel online softmax stats: max and 1/sumexp ----------
__global__ void k_stats(const float* __restrict__ logits,
                        float* __restrict__ maxv, float* __restrict__ invs) {
  int p = blockIdx.x * 256 + threadIdx.x;
  int n = p / HW, hw = p - n * HW;
  const float* base = logits + (size_t)n * NCH * HW + hw;
  float m = -INFINITY, s = 0.0f;
  for (int c = 0; c < NCH; ++c) {
    float l = base[(size_t)c * HW];
    float nm = fmaxf(m, l);
    s = s * expf(m - nm) + expf(l - nm);
    m = nm;
  }
  maxv[p] = m;
  invs[p] = 1.0f / s;
}

// ---------- build sortable records: (~err_bits)<<32 | fg ----------
__global__ void k_buildkeys(const float* __restrict__ logits,
                            const int* __restrict__ labels,
                            const float* __restrict__ maxv,
                            const float* __restrict__ invs,
                            u64* __restrict__ keys, int cls) {
  int p = blockIdx.x * 256 + threadIdx.x;
  int n = p / HW, hw = p - n * HW;
  int lab = labels[p];
  bool valid = (lab != IGNORE_IDX);
  u32 fg = (valid && lab == cls) ? 1u : 0u;
  float l = logits[((size_t)n * NCH + cls) * HW + hw];
  float prob = expf(l - maxv[p]) * invs[p];
  float err = valid ? (fg ? (1.0f - prob) : prob) : 0.0f;
  err = fmaxf(err, 0.0f);
  u32 key = ~__float_as_uint(err);  // ascending key sort == descending err
  keys[p] = ((u64)key << 32) | (u64)fg;
}

// ---------- radix pass 1: per-block 256-bin digit histogram ----------
__global__ void k_hist(const u64* __restrict__ src, u32* __restrict__ hist, int shift) {
  __shared__ u32 lh[256];
  int t = threadIdx.x;
  lh[t] = 0;
  __syncthreads();
  size_t base = (size_t)blockIdx.x * TILE;
#pragma unroll
  for (int r = 0; r < 8; ++r) {
    if (r < 7) __builtin_prefetch(&src[base + (size_t)(r + 1) * 256 + t], 0, 1);
    u64 rec = src[base + (size_t)r * 256 + t];
    atomicAdd(&lh[(u32)(rec >> shift) & 255u], 1u);
  }
  __syncthreads();
  hist[blockIdx.x * 256 + t] = lh[t];
}

// ---------- radix pass 2: in-place hist -> global scatter bases ----------
__global__ void k_scan(u32* __restrict__ hist) {
  __shared__ u32 tot[256];
  __shared__ u32 dstart[256];
  int d = threadIdx.x;
  u32 run = 0;
  for (int b = 0; b < NBLK; ++b) run += hist[b * 256 + d];
  tot[d] = run;
  __syncthreads();
  if (d == 0) {
    u32 a = 0;
    for (int i = 0; i < 256; ++i) { u32 h = tot[i]; dstart[i] = a; a += h; }
  }
  __syncthreads();
  u32 pos = dstart[d];
  for (int b = 0; b < NBLK; ++b) {
    u32 h = hist[b * 256 + d];
    hist[b * 256 + d] = pos;
    pos += h;
  }
}

// ---------- radix pass 3: stable scatter via wave32 ballot matching ----------
__global__ void k_scatter(const u64* __restrict__ src, u64* __restrict__ dst,
                          const u32* __restrict__ sbase, int shift) {
  __shared__ u32 running[256];
  __shared__ unsigned short wcnt[8 * 256];
  __shared__ u32 wbase[8 * 256];
  int t = threadIdx.x;
  int lane = t & 31, w = t >> 5;
  running[t] = sbase[blockIdx.x * 256 + t];
  size_t base = (size_t)blockIdx.x * TILE;
  __syncthreads();
  for (int r = 0; r < 8; ++r) {
    if (r < 7) __builtin_prefetch(&src[base + (size_t)(r + 1) * 256 + t], 0, 1);
    u64 rec = src[base + (size_t)r * 256 + t];
    u32 d = (u32)(rec >> shift) & 255u;
    u32 m = 0xffffffffu;
#pragma unroll
    for (int bit = 0; bit < 8; ++bit) {
      u32 bal = __builtin_amdgcn_ballot_w32((bool)((d >> bit) & 1u));
      m &= ((d >> bit) & 1u) ? bal : ~bal;
    }
    u32 rank = __popc(m & ((1u << lane) - 1u));
    u32 cnt  = __popc(m);
#pragma unroll
    for (int j = 0; j < 8; ++j) wcnt[j * 256 + t] = 0;
    __syncthreads();
    if (rank == 0) wcnt[w * 256 + d] = (unsigned short)cnt;
    __syncthreads();
    u32 bse = running[t];
#pragma unroll
    for (int w2 = 0; w2 < 8; ++w2) {
      wbase[w2 * 256 + t] = bse;
      bse += (u32)wcnt[w2 * 256 + t];
    }
    running[t] = bse;
    __syncthreads();
    dst[wbase[w * 256 + d] + rank] = rec;
  }
}

// ---------- per-block fg totals via WMMA (exact: fg in {0,1}, f32 acc) ----------
__global__ void k_fgsum(const u64* __restrict__ keys, u32* __restrict__ fgsum) {
  int lane = threadIdx.x;   // blockDim = 32, all lanes active (EXEC all 1s)
  v16h ones;
#pragma unroll
  for (int i = 0; i < 16; ++i) ones[i] = (_Float16)1.0f;
  v8f acc = {};
  size_t base = (size_t)blockIdx.x * TILE;
  for (int tile = 0; tile < 4; ++tile) {
    v16h a;
#pragma unroll
    for (int j = 0; j < 16; ++j) {
      u64 rec = keys[base + (size_t)tile * 512 + (size_t)j * 32 + lane];
      a[j] = (_Float16)(float)(u32)(rec & 1ull);
    }
    // D[m][n] = rowsum(A_m) with B = ones; sum over all entries = 16 * sum(A)
    acc = __builtin_amdgcn_wmma_f32_16x16x32_f16(false, a, false, ones,
                                                 (short)0, acc, false, false);
  }
  float s = 0.0f;
#pragma unroll
  for (int i = 0; i < 8; ++i) s += acc[i];
  __shared__ float red[32];
  red[lane] = s;
  __syncthreads();
  if (lane == 0) {
    float tot = 0.0f;
    for (int i = 0; i < 32; ++i) tot += red[i];
    fgsum[blockIdx.x] = (u32)(tot * (1.0f / 16.0f) + 0.5f);
  }
}

// ---------- exclusive scan of block fg sums + total (gts) ----------
__global__ void k_fgscan(const u32* __restrict__ fgsum, u32* __restrict__ fgbase,
                         u32* __restrict__ gts) {
  u32 run = 0;
  for (int b = 0; b < NBLK; ++b) { u32 h = fgsum[b]; fgbase[b] = run; run += h; }
  *gts = run;
}

// ---------- Lovasz loss accumulation over sorted order ----------
__global__ void k_loss(const u64* __restrict__ keys, const u32* __restrict__ fgbase,
                       const u32* __restrict__ gtsPtr, double* __restrict__ bpart) {
  __shared__ u32 runF;
  __shared__ u32 waveF[8];
  __shared__ u32 waveFB[8];
  __shared__ double red[256];
  int t = threadIdx.x;
  int lane = t & 31, w = t >> 5;
  if (t == 0) runF = fgbase[blockIdx.x];
  __syncthreads();
  float gtsf = (float)(*gtsPtr);
  size_t base = (size_t)blockIdx.x * TILE;
  double lsum = 0.0;
  for (int r = 0; r < 8; ++r) {
    size_t e = base + (size_t)r * 256 + t;     // global sorted index (< 2^24)
    u64 rec = keys[e];
    u32 fg = (u32)(rec & 1ull);
    u32 bal = __builtin_amdgcn_ballot_w32(fg != 0u);
    u32 pre = __popc(bal & ((1u << lane) - 1u));
    if (lane == 0) waveF[w] = __popc(bal);
    __syncthreads();
    if (t == 0) {
      u32 bse = runF;
#pragma unroll
      for (int w2 = 0; w2 < 8; ++w2) { waveFB[w2] = bse; bse += waveF[w2]; }
      runF = bse;
    }
    __syncthreads();
    u32 Fexcl = waveFB[w] + pre;               // cumF at i-1
    float err = __uint_as_float(~(u32)(rec >> 32));
    float Fi = (float)(Fexcl + fg);            // cumF at i
    float Fp = (float)Fexcl;
    float fi = (float)e;
    float ji   = 1.0f - (gtsf - Fi) / (gtsf + (fi + 1.0f) - Fi);
    float jim1 = (e == 0) ? 0.0f
                          : (1.0f - (gtsf - Fp) / (gtsf + fi - Fp));
    lsum += (double)(err * (ji - jim1));
  }
  red[t] = lsum;
  __syncthreads();
  for (int st = 128; st > 0; st >>= 1) {
    if (t < st) red[t] += red[t + st];
    __syncthreads();
  }
  if (t == 0) bpart[blockIdx.x] = red[0];
}

// ---------- deterministic reduce of block partials -> per-class loss ----------
__global__ void k_classred(const double* __restrict__ bpart, double* __restrict__ lossOut) {
  __shared__ double red[256];
  int t = threadIdx.x;
  double s = 0.0;
  for (int b = t; b < NBLK; b += 256) s += bpart[b];
  red[t] = s;
  __syncthreads();
  for (int st = 128; st > 0; st >>= 1) {
    if (t < st) red[t] += red[t + st];
    __syncthreads();
  }
  if (t == 0) *lossOut = red[0];
}

// ---------- mean over classes ----------
__global__ void k_final(const double* __restrict__ losses, float* __restrict__ out) {
  double s = 0.0;
  for (int c = 0; c < NCH; ++c) s += losses[c];
  out[0] = (float)(s / (double)NCH);
}

extern "C" void kernel_launch(void* const* d_in, const int* in_sizes, int n_in,
                              void* d_out, int out_size, void* d_ws, size_t ws_size,
                              hipStream_t stream) {
  (void)in_sizes; (void)n_in; (void)out_size; (void)ws_size;
  const float* logits = (const float*)d_in[0];
  const int*   labels = (const int*)d_in[1];
  float* out = (float*)d_out;
  char* ws = (char*)d_ws;

  size_t off = 0;
  auto wsAlloc = [&](size_t bytes) -> void* {
    off = (off + 255) & ~(size_t)255;
    void* p = ws + off;
    off += bytes;
    return p;
  };
  float* maxv   = (float*)wsAlloc((size_t)NPIX * 4);
  float* invs   = (float*)wsAlloc((size_t)NPIX * 4);
  u64*   keysA  = (u64*)  wsAlloc((size_t)NPIX * 8);
  u64*   keysB  = (u64*)  wsAlloc((size_t)NPIX * 8);
  u32*   hist   = (u32*)  wsAlloc((size_t)NBLK * 256 * 4);
  u32*   fgsum  = (u32*)  wsAlloc((size_t)NBLK * 4);
  u32*   fgbase = (u32*)  wsAlloc((size_t)NBLK * 4);
  u32*   gts    = (u32*)  wsAlloc(64);
  double* bpart = (double*)wsAlloc((size_t)NBLK * 8);
  double* losses= (double*)wsAlloc((size_t)NCH * 8);

  k_stats<<<NPIX / 256, 256, 0, stream>>>(logits, maxv, invs);
  for (int c = 0; c < NCH; ++c) {
    k_buildkeys<<<NPIX / 256, 256, 0, stream>>>(logits, labels, maxv, invs, keysA, c);
    u64* src = keysA;
    u64* dst = keysB;
    for (int pass = 0; pass < 4; ++pass) {
      int shift = 32 + 8 * pass;
      k_hist<<<NBLK, 256, 0, stream>>>(src, hist, shift);
      k_scan<<<1, 256, 0, stream>>>(hist);
      k_scatter<<<NBLK, 256, 0, stream>>>(src, dst, hist, shift);
      u64* tmp = src; src = dst; dst = tmp;
    }
    // after 4 passes, sorted data is back in keysA (src)
    k_fgsum<<<NBLK, 32, 0, stream>>>(src, fgsum);
    k_fgscan<<<1, 1, 0, stream>>>(fgsum, fgbase, gts);
    k_loss<<<NBLK, 256, 0, stream>>>(src, fgbase, gts, bpart);
    k_classred<<<1, 256, 0, stream>>>(bpart, losses + c);
  }
  k_final<<<1, 1, 0, stream>>>(losses, out);
}
